// PIGNN_SeparatedCoords_29669634081217
// MI455X (gfx1250) — matrix operand order repaired and checked by reading, not compiled, not run on hardware
//
#include <hip/hip_runtime.h>
#include <hip/hip_bf16.h>

typedef __attribute__((ext_vector_type(16))) __bf16 v16bf;
typedef __attribute__((ext_vector_type(8)))  __bf16 v8bf;
typedef __attribute__((ext_vector_type(4)))  __bf16 v4bf;
typedef __attribute__((ext_vector_type(8)))  float  v8f;

#define HDIM 128

// ---------------------------------------------------------------------------
// Fused GEMM:  out[M x NCOL] = act( A[M x 128] @ W[128 x NCOL] + bias + addend )
//   ABF=true  : A is bf16 in global (pure b128 copy into LDS)
//   ABF=false : A is fp32 (converted to bf16 during staging)
// Dual epilogue: fp32 `out` and/or bf16 `out_bf` (either may be null).
// v_wmma_f32_16x16x32_bf16, f32 accumulate.
// Block: 64 rows x NCOL cols, 256 threads (8 waves); waves 0..3 left col-half.
// LDS layouts make every fragment a contiguous vector load:
//   sA row-major  -> A frag = two 16B runs (2x ds_load_b128)
//   sWt col-major -> B frag = two 16B runs (2x ds_load_b128)
// ---------------------------------------------------------------------------
template <int NT, bool ABF>   // NT: 16-wide col tiles (8 -> NCOL=128, 4 -> 64)
__global__ __launch_bounds__(256) void gemm128(
    const void* __restrict__ Ap, const float* __restrict__ W,
    const float* __restrict__ bias, const float* __restrict__ addend,
    float* __restrict__ out, __bf16* __restrict__ out_bf, int M, int relu_flag)
{
    constexpr int NCOL   = NT * 16;
    constexpr int NTW    = NT / 2;
    constexpr int SA_LD  = HDIM + 8;    // 136 bf16: 16B-aligned, 4-bank lane stride
    constexpr int SWT_LD = HDIM + 8;

    __shared__ __bf16 sA [64   * SA_LD ];   // [row][k]
    __shared__ __bf16 sWt[NCOL * SWT_LD];   // [col][k] (transposed W)

    const int tid  = threadIdx.x;
    const int row0 = blockIdx.x * 64;

    // ---- stage A tile ----
    if (ABF) {
        const __bf16* A = (const __bf16*)Ap;
        for (int p = tid; p < 64 * 8; p += 256) {     // 16 bf16 (32B) per task
            int r = p >> 3, c16 = (p & 7) << 4;
            int gr = row0 + r;
            v8bf lo, hi;
            if (gr < M) {
                lo = *(const v8bf*)(A + (size_t)gr * HDIM + c16);
                hi = *(const v8bf*)(A + (size_t)gr * HDIM + c16 + 8);
            } else {
#pragma unroll
                for (int k = 0; k < 8; ++k) { lo[k] = (__bf16)0.f; hi[k] = (__bf16)0.f; }
            }
            *(v8bf*)(sA + r * SA_LD + c16)     = lo;
            *(v8bf*)(sA + r * SA_LD + c16 + 8) = hi;
        }
    } else {
        const float* A = (const float*)Ap;
        for (int p = tid; p < 64 * 16; p += 256) {    // 8 floats -> one v8bf
            int r = p >> 4, c8 = (p & 15) << 3;
            int gr = row0 + r;
            float4 lo = make_float4(0.f, 0.f, 0.f, 0.f), hi = lo;
            if (gr < M) {
                lo = *(const float4*)(A + (size_t)gr * HDIM + c8);
                hi = *(const float4*)(A + (size_t)gr * HDIM + c8 + 4);
            }
            v8bf v;
            v[0] = (__bf16)lo.x; v[1] = (__bf16)lo.y; v[2] = (__bf16)lo.z; v[3] = (__bf16)lo.w;
            v[4] = (__bf16)hi.x; v[5] = (__bf16)hi.y; v[6] = (__bf16)hi.z; v[7] = (__bf16)hi.w;
            *(v8bf*)(sA + r * SA_LD + c8) = v;
        }
    }
    // ---- stage W transposed: coalesced reads (lanes = cols) ----
    for (int p = tid; p < NCOL * 8; p += 256) {
        int c  = p & (NCOL - 1);
        int k0 = (p / NCOL) * 16;
        v8bf lo, hi;
#pragma unroll
        for (int k = 0; k < 8; ++k) {
            lo[k] = (__bf16)W[(size_t)(k0 + k)     * NCOL + c];
            hi[k] = (__bf16)W[(size_t)(k0 + k + 8) * NCOL + c];
        }
        *(v8bf*)(sWt + c * SWT_LD + k0)     = lo;
        *(v8bf*)(sWt + c * SWT_LD + k0 + 8) = hi;
    }
    __syncthreads();

    const int wave = tid >> 5;
    const int lane = tid & 31;
    const int half = lane >> 4;
    const int l16  = lane & 15;
    const int rw   = wave & 3;
    const int cg   = wave >> 2;
    const int arow = rw * 16 + l16;

    union Frag { v16bf v; v8bf c[2]; };

    v8f acc[NTW];
#pragma unroll
    for (int nt = 0; nt < NTW; ++nt)
#pragma unroll
        for (int r = 0; r < 8; ++r) acc[nt][r] = 0.f;

#pragma unroll
    for (int kk = 0; kk < 4; ++kk) {
        const int kb = kk * 32;
        Frag af;
        af.c[0] = *(const v8bf*)(sA + arow * SA_LD + kb + half * 8);
        af.c[1] = *(const v8bf*)(sA + arow * SA_LD + kb + 16 + half * 8);
#pragma unroll
        for (int nt = 0; nt < NTW; ++nt) {
            const int col = (cg * NTW + nt) * 16 + l16;
            Frag bf;
            bf.c[0] = *(const v8bf*)(sWt + col * SWT_LD + kb + half * 8);
            bf.c[1] = *(const v8bf*)(sWt + col * SWT_LD + kb + 16 + half * 8);
            acc[nt] = __builtin_amdgcn_wmma_f32_16x16x32_bf16(
                false, af.v, false, bf.v, (short)0, acc[nt], false, false);
        }
    }

    // ---- epilogue: C layout VGPR r -> rows r / r+8 (lane half) ----
#pragma unroll
    for (int nt = 0; nt < NTW; ++nt) {
        const int col = (cg * NTW + nt) * 16 + l16;
#pragma unroll
        for (int r = 0; r < 8; ++r) {
            int grow = row0 + rw * 16 + half * 8 + r;
            if (grow < M) {
                float vv = acc[nt][r];
                if (bias)      vv += bias[col];
                if (addend)    vv += addend[(size_t)grow * NCOL + col];
                if (relu_flag) vv = fmaxf(vv, 0.f);
                if (out)    out   [(size_t)grow * NCOL + col] = vv;
                if (out_bf) out_bf[(size_t)grow * NCOL + col] = (__bf16)vv;
            }
        }
    }
}

// ---------------------------------------------------------------------------
// Small-K first layers: out_bf[M x 128] = relu(X[:, koff:koff+K] @ W1 + b1)
// 4 output features per thread, v4bf stores.
// ---------------------------------------------------------------------------
__global__ __launch_bounds__(256) void enc_first_kernel(
    const float* __restrict__ X, int ldx, int koff, int K,
    const float* __restrict__ W1, const float* __restrict__ b1,
    __bf16* __restrict__ out, int M)
{
    int idx = blockIdx.x * blockDim.x + threadIdx.x;
    if (idx >= M * 32) return;
    int n = idx >> 5, j4 = (idx & 31) << 2;
    float acc[4];
#pragma unroll
    for (int j = 0; j < 4; ++j) acc[j] = b1[j4 + j];
    for (int k = 0; k < K; ++k) {
        float xv = X[(size_t)n * ldx + koff + k];
#pragma unroll
        for (int j = 0; j < 4; ++j)
            acc[j] = fmaf(xv, W1[k * HDIM + j4 + j], acc[j]);
    }
    v4bf o;
#pragma unroll
    for (int j = 0; j < 4; ++j) o[j] = (__bf16)fmaxf(acc[j], 0.f);
    *(v4bf*)(out + (size_t)n * HDIM + j4) = o;
}

// m1 = relu(a[src] + b[dst] + eW + be1)   (all bf16, 4 feats/thread)
__global__ __launch_bounds__(256) void edge_combine_kernel(
    const __bf16* __restrict__ a, const __bf16* __restrict__ b,
    const __bf16* __restrict__ eW, const float* __restrict__ be1,
    const int* __restrict__ src, const int* __restrict__ dst,
    __bf16* __restrict__ out, int E)
{
    int idx = blockIdx.x * blockDim.x + threadIdx.x;
    if (idx >= E * 32) return;
    int e = idx >> 5, f4 = (idx & 31) << 2;
    int s = src[e], d = dst[e];
    v4bf va = *(const v4bf*)(a  + (size_t)s * HDIM + f4);
    v4bf vb = *(const v4bf*)(b  + (size_t)d * HDIM + f4);
    v4bf ve = *(const v4bf*)(eW + (size_t)e * HDIM + f4);
    v4bf o;
#pragma unroll
    for (int k = 0; k < 4; ++k) {
        float v = (float)va[k] + (float)vb[k] + (float)ve[k] + be1[f4 + k];
        o[k] = (__bf16)fmaxf(v, 0.f);
    }
    *(v4bf*)(out + (size_t)e * HDIM + f4) = o;
}

// agg[dst[e]] += m[e]   (f32 atomics, 4 feats/thread, one index load)
__global__ __launch_bounds__(256) void scatter_add_kernel(
    const float* __restrict__ m, const int* __restrict__ dst,
    float* __restrict__ agg, int E)
{
    int idx = blockIdx.x * blockDim.x + threadIdx.x;
    if (idx >= E * 32) return;
    int e = idx >> 5, f4 = (idx & 31) << 2;
    float4 v = *(const float4*)(m + (size_t)e * HDIM + f4);
    float* base = agg + (size_t)dst[e] * HDIM + f4;
    atomicAdd(base + 0, v.x);
    atomicAdd(base + 1, v.y);
    atomicAdd(base + 2, v.z);
    atomicAdd(base + 3, v.w);
}

__global__ __launch_bounds__(256) void fill_zero_kernel(float4* __restrict__ p, int n4)
{
    int i = blockIdx.x * blockDim.x + threadIdx.x;
    if (i < n4) p[i] = make_float4(0.f, 0.f, 0.f, 0.f);
}

// cc[n][j] = coords[n][0] * dec_W1[128][j] + coords[n][2] * dec_W1[129][j]
__global__ __launch_bounds__(256) void coord_contrib_kernel(
    const float* __restrict__ coords, const float* __restrict__ decW1,
    float* __restrict__ cc, int N)
{
    int idx = blockIdx.x * blockDim.x + threadIdx.x;
    if (idx >= N * HDIM) return;
    int n = idx >> 7, j = idx & 127;
    cc[idx] = coords[(size_t)n * 3 + 0] * decW1[128 * HDIM + j]
            + coords[(size_t)n * 3 + 2] * decW1[129 * HDIM + j];
}

// pred = (z2 @ W3 + b3) with displacement/rotation masks
__global__ __launch_bounds__(256) void decoder_final_kernel(
    const float* __restrict__ z2, const float* __restrict__ W3,
    const float* __restrict__ b3, const float* __restrict__ bc_disp,
    const float* __restrict__ bc_rot, float* __restrict__ out, int N)
{
    int n = blockIdx.x * blockDim.x + threadIdx.x;
    if (n >= N) return;
    const float* z = z2 + (size_t)n * 64;
    float p0 = b3[0], p1 = b3[1], p2 = b3[2];
    for (int k = 0; k < 64; ++k) {
        float zv = z[k];
        p0 = fmaf(zv, W3[k * 3 + 0], p0);
        p1 = fmaf(zv, W3[k * 3 + 1], p1);
        p2 = fmaf(zv, W3[k * 3 + 2], p2);
    }
    float dm = 1.f - bc_disp[n];
    float rm = 1.f - bc_rot[n];
    out[(size_t)n * 3 + 0] = p0 * dm;
    out[(size_t)n * 3 + 1] = p1 * dm;
    out[(size_t)n * 3 + 2] = p2 * rm;
}

// ---------------------------------------------------------------------------
extern "C" void kernel_launch(void* const* d_in, const int* in_sizes, int n_in,
                              void* d_out, int out_size, void* d_ws, size_t ws_size,
                              hipStream_t stream)
{
    const float* x         = (const float*)d_in[0];
    const float* coords    = (const float*)d_in[1];
    const float* edge_attr = (const float*)d_in[2];
    const float* bc_disp   = (const float*)d_in[3];
    const float* bc_rot    = (const float*)d_in[4];
    const int*   ei        = (const int*)  d_in[5];
    const float* enc_n_W1  = (const float*)d_in[6];
    const float* enc_n_b1  = (const float*)d_in[7];
    const float* enc_n_W2  = (const float*)d_in[8];
    const float* enc_n_b2  = (const float*)d_in[9];
    const float* enc_e_W1  = (const float*)d_in[10];
    const float* enc_e_b1  = (const float*)d_in[11];
    const float* enc_e_W2  = (const float*)d_in[12];
    const float* enc_e_b2  = (const float*)d_in[13];
    const float* mp_Ws     = (const float*)d_in[14];
    const float* mp_Wd     = (const float*)d_in[15];
    const float* mp_We     = (const float*)d_in[16];
    const float* mp_be1    = (const float*)d_in[17];
    const float* mp_W2     = (const float*)d_in[18];
    const float* mp_b2     = (const float*)d_in[19];
    const float* mp_Wn_h   = (const float*)d_in[20];
    const float* mp_Wn_a   = (const float*)d_in[21];
    const float* mp_bn1    = (const float*)d_in[22];
    const float* mp_Wn2    = (const float*)d_in[23];
    const float* mp_bn2    = (const float*)d_in[24];
    const float* dec_W1    = (const float*)d_in[25];
    const float* dec_b1    = (const float*)d_in[26];
    const float* dec_W2    = (const float*)d_in[27];
    const float* dec_b2    = (const float*)d_in[28];
    const float* dec_W3    = (const float*)d_in[29];
    const float* dec_b3    = (const float*)d_in[30];

    const int N = in_sizes[1] / 3;
    const int E = in_sizes[2] / 10;
    const int L = in_sizes[14] / (HDIM * HDIM);
    const int* src = ei;
    const int* dst = ei + E;

    // ---- workspace layout: bump allocator, 256B aligned ----
    const size_t EH = (size_t)E * HDIM, NH = (size_t)N * HDIM;
    char* base = (char*)d_ws;
    auto alloc = [&](size_t bytes) {
        char* p = base;
        base += (bytes + 255) & ~(size_t)255;
        return p;
    };
    __bf16* e_bf   = (__bf16*)alloc(EH * 2);   // edge features, persists
    __bf16* et1_bf = (__bf16*)alloc(EH * 2);   // edge enc tmp / eW scratch
    __bf16* et2_bf = (__bf16*)alloc(EH * 2);   // m1 scratch
    float*  mfp    = (float*) alloc(EH * 4);   // pre-scatter messages (fp32)
    float*  h      = (float*) alloc(NH * 4);   // node state fp32
    __bf16* h_bf   = (__bf16*)alloc(NH * 2);   // bf16 shadow of h
    __bf16* av_bf  = (__bf16*)alloc(NH * 2);   // a = h@Ws (also enc tmp, z1)
    __bf16* bv_bf  = (__bf16*)alloc(NH * 2);   // b = h@Wd
    __bf16* t2_bf  = (__bf16*)alloc(NH * 2);   // pre-update hidden
    float*  agg    = (float*) alloc(NH * 4);   // segment sums (atomics)
    float*  t1f    = (float*) alloc(NH * 4);   // h@Wn_h / coord contrib / z2
    (void)ws_size; (void)n_in; (void)out_size;

    const int gN  = (N + 63) / 64;                 // GEMM blocks over nodes
    const int gE  = (E + 63) / 64;                 // GEMM blocks over edges
    const int qN  = (N * 32 + 255) / 256;          // 4-feature elementwise, nodes
    const int qE  = (E * 32 + 255) / 256;          // 4-feature elementwise, edges
    const int eNH = ((int)NH + 255) / 256;

    // ---- encoders ----
    enc_first_kernel<<<qN, 256, 0, stream>>>(x, 9, 3, 6, enc_n_W1, enc_n_b1, av_bf, N);
    gemm128<8, true><<<gN, 256, 0, stream>>>(av_bf, enc_n_W2, enc_n_b2, nullptr,
                                             h, h_bf, N, 0);
    enc_first_kernel<<<qE, 256, 0, stream>>>(edge_attr, 10, 0, 10, enc_e_W1, enc_e_b1,
                                             et1_bf, E);
    gemm128<8, true><<<gE, 256, 0, stream>>>(et1_bf, enc_e_W2, enc_e_b2, nullptr,
                                             nullptr, e_bf, E, 0);

    // ---- message passing layers ----
    for (int i = 0; i < L; ++i) {
        const size_t wo = (size_t)i * HDIM * HDIM;
        const size_t bo = (size_t)i * HDIM;
        gemm128<8, true><<<gN, 256, 0, stream>>>(h_bf, mp_Ws + wo, nullptr, nullptr,
                                                 nullptr, av_bf, N, 0);
        gemm128<8, true><<<gN, 256, 0, stream>>>(h_bf, mp_Wd + wo, nullptr, nullptr,
                                                 nullptr, bv_bf, N, 0);
        gemm128<8, true><<<gE, 256, 0, stream>>>(e_bf, mp_We + wo, nullptr, nullptr,
                                                 nullptr, et1_bf, E, 0);
        edge_combine_kernel<<<qE, 256, 0, stream>>>(av_bf, bv_bf, et1_bf, mp_be1 + bo,
                                                    src, dst, et2_bf, E);
        gemm128<8, true><<<gE, 256, 0, stream>>>(et2_bf, mp_W2 + wo, mp_b2 + bo, nullptr,
                                                 mfp, nullptr, E, 0);
        fill_zero_kernel<<<(eNH + 3) / 4 * 1 + ((int)(NH / 4) + 255) / 256 * 0 +
                           ((int)(NH / 4) + 255) / 256, 256, 0, stream>>>(
            (float4*)agg, (int)(NH / 4));
        scatter_add_kernel<<<qE, 256, 0, stream>>>(mfp, dst, agg, E);
        gemm128<8, true><<<gN, 256, 0, stream>>>(h_bf, mp_Wn_h + wo, nullptr, nullptr,
                                                 t1f, nullptr, N, 0);
        gemm128<8, false><<<gN, 256, 0, stream>>>(agg, mp_Wn_a + wo, mp_bn1 + bo, t1f,
                                                  nullptr, t2_bf, N, 1);
        gemm128<8, true><<<gN, 256, 0, stream>>>(t2_bf, mp_Wn2 + wo, mp_bn2 + bo, h,
                                                 h, h_bf, N, 0);   // h += upd
    }

    // ---- decoder ----
    coord_contrib_kernel<<<eNH, 256, 0, stream>>>(coords, dec_W1, t1f, N);
    gemm128<8, true><<<gN, 256, 0, stream>>>(h_bf, dec_W1, dec_b1, t1f,
                                             nullptr, av_bf, N, 1);  // z1 (130->128 split)
    gemm128<4, true><<<gN, 256, 0, stream>>>(av_bf, dec_W2, dec_b2, nullptr,
                                             t1f, nullptr, N, 1);    // z2: N x 64 fp32
    decoder_final_kernel<<<(N + 255) / 256, 256, 0, stream>>>(
        t1f, dec_W3, dec_b3, bc_disp, bc_rot, (float*)d_out, N);
}
